// PatchConv2LayerClassifier_8117488190078
// MI455X (gfx1250) — matrix operand order repaired and checked by reading, not compiled, not run on hardware
//
#include <hip/hip_runtime.h>
#include <hip/hip_bf16.h>

#define N_NODES 131072
#define N_EDGES 1048576
#define IN_F    64
#define HID     256
#define RD      128
#define NPG     128
#define N_GRAPHS (N_NODES / NPG)

typedef __bf16 bf16_t;
typedef __attribute__((ext_vector_type(16))) __bf16 v16bf;
typedef __attribute__((ext_vector_type(8)))  __bf16 v8bf;
typedef __attribute__((ext_vector_type(8)))  float  v8f;

union V16U { v16bf v; v8bf h[2]; };

__device__ __forceinline__ float leaky_f(float x) { return x >= 0.f ? x : 0.01f * x; }

// ---------------- utility: zero fill (graph-capture safe) ----------------
__global__ void k_zero(float4* __restrict__ p, unsigned n4) {
    unsigned i = blockIdx.x * blockDim.x + threadIdx.x;
    if (i < n4) { float4 z = {0.f, 0.f, 0.f, 0.f}; p[i] = z; }
}

// ---------------- structural degrees ----------------
__global__ void k_count(const int* __restrict__ src, const int* __restrict__ dst,
                        unsigned* __restrict__ cs, unsigned* __restrict__ cd) {
    unsigned e = blockIdx.x * blockDim.x + threadIdx.x;
    if (e < N_EDGES) {
        atomicAdd(&cs[src[e]], 1u);
        atomicAdd(&cd[dst[e]], 1u);
    }
}

__global__ void k_invdeg(const unsigned* __restrict__ cs, const unsigned* __restrict__ cd,
                         float* __restrict__ inv_out, float* __restrict__ inv_in) {
    unsigned i = blockIdx.x * blockDim.x + threadIdx.x;
    if (i < N_NODES) {
        unsigned a = cs[i] ? cs[i] : 1u;
        unsigned b = cd[i] ? cd[i] : 1u;
        inv_out[i] = rsqrtf((float)a);
        inv_in[i]  = rsqrtf((float)b);
    }
}

// ---------------- weight prep: fp32 row-major -> bf16 N-major (K contiguous) ----------------
__global__ void k_prepw(const float* __restrict__ W1, const float* __restrict__ W2,
                        bf16_t* __restrict__ W1t, bf16_t* __restrict__ W2t) {
    unsigned i = blockIdx.x * blockDim.x + threadIdx.x;
    if (i < IN_F * HID) {                       // W1: 64x256 -> W1t[n][k], n in [0,256)
        unsigned n = i / IN_F, k = i % IN_F;
        W1t[i] = (bf16_t)W1[k * HID + n];
    } else if (i < IN_F * HID + HID * RD) {     // W2: 256x128 -> W2t[n][k], n in [0,128)
        unsigned j = i - IN_F * HID;
        unsigned n = j / HID, k = j % HID;
        W2t[j] = (bf16_t)W2[k * RD + n];
    }
}

// ---------------- layer-1 edge scatter: agg1[dst] += feat[src] * (w_e * deg_out^-1/2) ----------------
__global__ void k_edge1(const float4* __restrict__ x4, const float* __restrict__ w,
                        const float* __restrict__ inv_out,
                        const int* __restrict__ src, const int* __restrict__ dst,
                        float* __restrict__ agg1) {
    unsigned t = blockIdx.x * blockDim.x + threadIdx.x;   // E*16 threads exactly
    unsigned e = t >> 4, c = t & 15;
    int s = src[e], d = dst[e];
    float sc = w[e] * inv_out[s];
    float4 v = x4[(size_t)s * 16 + c];
    float* ap = agg1 + (size_t)d * IN_F + c * 4;
    atomicAdd(ap + 0, v.x * sc);
    atomicAdd(ap + 1, v.y * sc);
    atomicAdd(ap + 2, v.z * sc);
    atomicAdd(ap + 3, v.w * sc);
}

// ---------------- GEMM1 (WMMA bf16): h1 = bf16( leaky((agg1 @ W1) * din) * dout ), N x 256 ----------------
// wave computes a 16x64 tile; K = 64 -> 2 wmma per 16x16 tile
__global__ void __launch_bounds__(256) k_gemm1(const float* __restrict__ agg1,
                                               const bf16_t* __restrict__ W1t,
                                               const float* __restrict__ inv_in,
                                               const float* __restrict__ inv_out,
                                               bf16_t* __restrict__ h1s) {
    unsigned wid  = (blockIdx.x * blockDim.x + threadIdx.x) >> 5;
    unsigned lane = threadIdx.x & 31;
    unsigned half = lane >> 4, r = lane & 15;
    unsigned mbase = (wid >> 2) * 16;
    unsigned nbase = (wid & 3) * 64;

    // A fragments in contiguous K order (matches B): lane<16 holds K {0..15}, lane>=16 K {16..31}
    const float* ap = agg1 + (size_t)(mbase + r) * IN_F;
    V16U a0, a1;
#pragma unroll
    for (int j = 0; j < 8; ++j) {
        a0.v[j]     = (bf16_t)ap[16 * half + j];
        a0.v[8 + j] = (bf16_t)ap[16 * half + 8 + j];
        a1.v[j]     = (bf16_t)ap[32 + 16 * half + j];
        a1.v[8 + j] = (bf16_t)ap[32 + 16 * half + 8 + j];
    }
    float din[8], dou[8];
#pragma unroll
    for (int i = 0; i < 8; ++i) {
        unsigned rr = mbase + half * 8 + i;   // C/D row layout: lanes16-31 hold M=8..15
        din[i] = inv_in[rr];
        dou[i] = inv_out[rr];
    }
#pragma unroll
    for (int nt = 0; nt < 4; ++nt) {
        unsigned col = nbase + nt * 16 + r;
        const bf16_t* bp = W1t + (size_t)col * IN_F + 16 * half;  // K-contiguous column
        v16bf b0 = *(const v16bf*)bp;
        v16bf b1 = *(const v16bf*)(bp + 32);
        v8f c = {};
        c = __builtin_amdgcn_wmma_f32_16x16x32_bf16(false, a0.v, false, b0, (short)0, c, false, false);
        c = __builtin_amdgcn_wmma_f32_16x16x32_bf16(false, a1.v, false, b1, (short)0, c, false, false);
#pragma unroll
        for (int i = 0; i < 8; ++i) {
            unsigned rr = mbase + half * 8 + i;
            float v = leaky_f(c[i] * din[i]) * dou[i];   // fold next layer's deg_out^-1/2
            h1s[(size_t)rr * HID + col] = (bf16_t)v;
        }
    }
}

// ---------------- GEMM2 (WMMA bf16): x2 = h1s @ W2, N x 128 fp32 ----------------
// wave computes a 16x64 tile; K = 256 -> 8 k-steps
__global__ void __launch_bounds__(256) k_gemm2(const bf16_t* __restrict__ h1s,
                                               const bf16_t* __restrict__ W2t,
                                               float* __restrict__ x2) {
    unsigned wid  = (blockIdx.x * blockDim.x + threadIdx.x) >> 5;
    unsigned lane = threadIdx.x & 31;
    unsigned half = lane >> 4, r = lane & 15;
    unsigned mbase = (wid >> 1) * 16;
    unsigned nbase = (wid & 1) * 64;

    const bf16_t* arow = h1s + (size_t)(mbase + r) * HID;
    v8f c0 = {}, c1 = {}, c2 = {}, c3 = {};
#pragma unroll
    for (int kb = 0; kb < HID; kb += 32) {
        V16U a;
        a.h[0] = *(const v8bf*)(arow + kb + 16 * half);
        a.h[1] = *(const v8bf*)(arow + kb + 16 * half + 8);
        const bf16_t* bb = W2t + kb + 16 * half;
        v16bf b;
        b  = *(const v16bf*)(bb + (size_t)(nbase + 0 * 16 + r) * HID);
        c0 = __builtin_amdgcn_wmma_f32_16x16x32_bf16(false, a.v, false, b, (short)0, c0, false, false);
        b  = *(const v16bf*)(bb + (size_t)(nbase + 1 * 16 + r) * HID);
        c1 = __builtin_amdgcn_wmma_f32_16x16x32_bf16(false, a.v, false, b, (short)0, c1, false, false);
        b  = *(const v16bf*)(bb + (size_t)(nbase + 2 * 16 + r) * HID);
        c2 = __builtin_amdgcn_wmma_f32_16x16x32_bf16(false, a.v, false, b, (short)0, c2, false, false);
        b  = *(const v16bf*)(bb + (size_t)(nbase + 3 * 16 + r) * HID);
        c3 = __builtin_amdgcn_wmma_f32_16x16x32_bf16(false, a.v, false, b, (short)0, c3, false, false);
    }
#pragma unroll
    for (int i = 0; i < 8; ++i) {
        size_t rr = (size_t)(mbase + half * 8 + i) * RD;
        x2[rr + nbase + 0 * 16 + r] = c0[i];
        x2[rr + nbase + 1 * 16 + r] = c1[i];
        x2[rr + nbase + 2 * 16 + r] = c2[i];
        x2[rr + nbase + 3 * 16 + r] = c3[i];
    }
}

// ---------------- layer-2 edge scatter: agg2[dst] += x2[src] * w_e ----------------
__global__ void k_edge2(const float4* __restrict__ x4, const float* __restrict__ w,
                        const int* __restrict__ src, const int* __restrict__ dst,
                        float* __restrict__ agg2) {
    unsigned t = blockIdx.x * blockDim.x + threadIdx.x;   // E*32 threads exactly
    unsigned e = t >> 5, c = t & 31;
    int s = src[e], d = dst[e];
    float sc = w[e];
    float4 v = x4[(size_t)s * 32 + c];
    float* ap = agg2 + (size_t)d * RD + c * 4;
    atomicAdd(ap + 0, v.x * sc);
    atomicAdd(ap + 1, v.y * sc);
    atomicAdd(ap + 2, v.z * sc);
    atomicAdd(ap + 3, v.w * sc);
}

// ---------------- pool (mean over 128 nodes, with din scale + leaky) + 128->64->16 head ----------------
__global__ void __launch_bounds__(128) k_pool_head(const float* __restrict__ agg2,
                                                   const float* __restrict__ inv_in,
                                                   const float* __restrict__ Wlin,
                                                   const float* __restrict__ Wcls,
                                                   float* __restrict__ out) {
    __shared__ float pooled[RD];
    __shared__ float tvec[64];
    unsigned g = blockIdx.x, d = threadIdx.x;
    const float* base = agg2 + (size_t)g * NPG * RD;
    float s = 0.f;
    for (int i = 0; i < NPG; ++i) {
        float din = inv_in[g * NPG + i];
        s += leaky_f(base[(size_t)i * RD + d] * din);
    }
    pooled[d] = s * (1.0f / NPG);
    __syncthreads();
    if (d < 64) {
        float acc = 0.f;
        for (int k = 0; k < RD; ++k) acc += pooled[k] * Wlin[k * 64 + d];
        tvec[d] = acc;
    }
    __syncthreads();
    if (d < 16) {
        float acc = 0.f;
        for (int j = 0; j < 64; ++j) acc += tvec[j] * Wcls[j * 16 + d];
        out[g * 16 + d] = acc;
    }
}

extern "C" void kernel_launch(void* const* d_in, const int* in_sizes, int n_in,
                              void* d_out, int out_size, void* d_ws, size_t ws_size,
                              hipStream_t stream) {
    (void)in_sizes; (void)n_in; (void)out_size; (void)ws_size;
    const float* node_feats = (const float*)d_in[0];
    const float* edge_w     = (const float*)d_in[1];
    const float* W1         = (const float*)d_in[2];
    const float* W2         = (const float*)d_in[3];
    const float* Wlin       = (const float*)d_in[4];
    const float* Wcls       = (const float*)d_in[5];
    const int*   esrc       = (const int*)d_in[6];
    const int*   edst       = (const int*)d_in[7];

    char* ws = (char*)d_ws;
    // [0, 2MB): counts + inv-degrees
    unsigned* cnt_out = (unsigned*)(ws + 0);
    unsigned* cnt_in  = (unsigned*)(ws + (size_t)N_NODES * 4);
    float*    inv_out = (float*)(ws + (size_t)N_NODES * 8);
    float*    inv_in  = (float*)(ws + (size_t)N_NODES * 12);
    // [2MB, 4MB): bf16 weights (K-contiguous)
    bf16_t* W1t = (bf16_t*)(ws + (2u << 20));
    bf16_t* W2t = (bf16_t*)(ws + (2u << 20) + (256u << 10));
    // R1 [4MB, 68MB): agg1 (32MB) then reused as x2 (64MB)
    // R2 [68MB, 132MB): h1s bf16 (64MB) then reused as agg2 (64MB)
    size_t R1 = (size_t)4 << 20;
    size_t R2 = R1 + (size_t)N_NODES * RD * 4;
    float*  agg1 = (float*)(ws + R1);
    float*  x2   = (float*)(ws + R1);
    bf16_t* h1s  = (bf16_t*)(ws + R2);
    float*  agg2 = (float*)(ws + R2);

    const unsigned TB = 256;

    // degrees
    k_zero<<<(N_NODES / 2) / TB, TB, 0, stream>>>((float4*)cnt_out, N_NODES / 2);
    k_count<<<N_EDGES / TB, TB, 0, stream>>>(esrc, edst, cnt_out, cnt_in);
    k_invdeg<<<N_NODES / TB, TB, 0, stream>>>(cnt_out, cnt_in, inv_out, inv_in);
    // weights
    k_prepw<<<(IN_F * HID + HID * RD + TB - 1) / TB, TB, 0, stream>>>(W1, W2, W1t, W2t);
    // layer 1: aggregate(64d) then project with WMMA
    k_zero<<<(N_NODES * IN_F / 4) / TB, TB, 0, stream>>>((float4*)agg1, N_NODES * IN_F / 4);
    k_edge1<<<(N_EDGES * 16) / TB, TB, 0, stream>>>((const float4*)node_feats, edge_w,
                                                    inv_out, esrc, edst, agg1);
    k_gemm1<<<(N_NODES / 16) * 4 * 32 / TB, TB, 0, stream>>>(agg1, W1t, inv_in, inv_out, h1s);
    // layer 2: project with WMMA then aggregate(128d)
    k_gemm2<<<(N_NODES / 16) * 2 * 32 / TB, TB, 0, stream>>>(h1s, W2t, x2);
    k_zero<<<((unsigned)(N_NODES * (RD / 4))) / TB, TB, 0, stream>>>((float4*)agg2,
                                                                    N_NODES * (RD / 4));
    k_edge2<<<(N_EDGES / TB) * 32, TB, 0, stream>>>((const float4*)x2, edge_w, esrc, edst, agg2);
    // readout + head
    k_pool_head<<<N_GRAPHS, 128, 0, stream>>>(agg2, inv_in, Wlin, Wcls, (float*)d_out);
}
